// DoubleAttention_14139032338691
// MI455X (gfx1250) — matrix operand order repaired and checked by reading, not compiled, not run on hardware
//
#include <hip/hip_runtime.h>
#include <hip/hip_bf16.h>

typedef __attribute__((ext_vector_type(16))) __bf16 bf16x16;
typedef __attribute__((ext_vector_type(8)))  float  f32x8;

union ABFrag { bf16x16 v; unsigned int u[8]; };

__device__ __forceinline__ unsigned short f2bf(float f) {
  unsigned int u = __float_as_uint(f);
  u += 0x7FFFu + ((u >> 16) & 1u);          // round-to-nearest-even
  return (unsigned short)(u >> 16);
}
__device__ __forceinline__ unsigned int pack2bf(float x, float y) {
  return (unsigned int)f2bf(x) | ((unsigned int)f2bf(y) << 16);
}

// ---------------------------------------------------------------------------
// C[b, M, N] (f32) = A[b, M, K] (f32 -> bf16) * W[N, K]^T (f32 -> bf16)
// Block: 256 threads = 8 waves; block tile 128(M) x 64(N); K-step 32.
// Each wave owns a 32x32 tile = 2x2 WMMA tiles -> 4 v_wmma per K-step with
// A/B fragment reuse.  Next K-tile is prefetched (global_prefetch_b8).
// ---------------------------------------------------------------------------
__launch_bounds__(256)
__global__ void gemm_bf16_wmma(const float* __restrict__ A, long strideAb, int lda,
                               const float* __restrict__ W, int ldw,
                               float* __restrict__ C, long strideCb, int ldc,
                               int K)
{
  __shared__ unsigned short As[128 * 32];
  __shared__ unsigned short Ws[64 * 32];
  const int n0 = blockIdx.x * 64;
  const int m0 = blockIdx.y * 128;
  const float* Ab = A + (long)blockIdx.z * strideAb;
  float*       Cb = C + (long)blockIdx.z * strideCb;

  const int t    = threadIdx.x;
  const int lane = t & 31, lm = lane & 15, half = lane >> 4;
  const int wave = t >> 5, wm = wave >> 1, wn = wave & 1;
  const int arow = t >> 1, ac0 = (t & 1) * 16;  // A staging: 128 rows x 32 k, 16 f32/thread
  const int wrow = t >> 2, wc0 = (t & 3) * 8;   // W staging:  64 rows x 32 k,  8 f32/thread

  f32x8 acc[2][2];
#pragma unroll
  for (int i = 0; i < 2; ++i)
#pragma unroll
    for (int j = 0; j < 2; ++j)
      acc[i][j] = (f32x8){0.f, 0.f, 0.f, 0.f, 0.f, 0.f, 0.f, 0.f};

  for (int kt = 0; kt < K; kt += 32) {
    const float* ap = Ab + (long)(m0 + arow) * lda + kt + ac0;
    const float* wp = W + (long)(n0 + wrow) * ldw + kt + wc0;
    if (kt + 32 < K) {                       // warm L2 for the next K tile
      __builtin_prefetch(ap + 32, 0, 0);
      __builtin_prefetch(wp + 32, 0, 0);
    }
    float4 a0 = *(const float4*)ap;
    float4 a1 = *(const float4*)(ap + 4);
    float4 a2 = *(const float4*)(ap + 8);
    float4 a3 = *(const float4*)(ap + 12);
    unsigned short* as = &As[arow * 32 + ac0];
    as[0]  = f2bf(a0.x); as[1]  = f2bf(a0.y); as[2]  = f2bf(a0.z); as[3]  = f2bf(a0.w);
    as[4]  = f2bf(a1.x); as[5]  = f2bf(a1.y); as[6]  = f2bf(a1.z); as[7]  = f2bf(a1.w);
    as[8]  = f2bf(a2.x); as[9]  = f2bf(a2.y); as[10] = f2bf(a2.z); as[11] = f2bf(a2.w);
    as[12] = f2bf(a3.x); as[13] = f2bf(a3.y); as[14] = f2bf(a3.z); as[15] = f2bf(a3.w);

    float4 w0 = *(const float4*)wp;
    float4 w1 = *(const float4*)(wp + 4);
    unsigned short* ws = &Ws[wrow * 32 + wc0];
    ws[0] = f2bf(w0.x); ws[1] = f2bf(w0.y); ws[2] = f2bf(w0.z); ws[3] = f2bf(w0.w);
    ws[4] = f2bf(w1.x); ws[5] = f2bf(w1.y); ws[6] = f2bf(w1.z); ws[7] = f2bf(w1.w);
    __syncthreads();

    ABFrag af[2], bfr[2];
#pragma unroll
    for (int i = 0; i < 2; ++i) {
      const int mrow = (wm * 32 + i * 16 + lm) * 32;
      const int nrow = (wn * 32 + i * 16 + lm) * 32;
#pragma unroll
      for (int v = 0; v < 8; ++v) {
        const int ka = ((v < 4) ? 2 * v : 16 + 2 * (v - 4)) + 8 * half;  // A 16-bit layout
        af[i].u[v]  = *(const unsigned int*)&As[mrow + ka];
        const int kb = ((v < 4) ? 2 * v : 16 + 2 * (v - 4)) + 8 * half;  // B 16-bit layout
        bfr[i].u[v] = *(const unsigned int*)&Ws[nrow + kb];
      }
    }
#pragma unroll
    for (int i = 0; i < 2; ++i)
#pragma unroll
      for (int j = 0; j < 2; ++j)
        acc[i][j] = __builtin_amdgcn_wmma_f32_16x16x32_bf16(false, af[i].v, false, bfr[j].v,
                                                            (short)0, acc[i][j], false, false);
    __syncthreads();
  }

  // C/D layout: row m = r + 8*half, col n = lane&15
#pragma unroll
  for (int i = 0; i < 2; ++i)
#pragma unroll
    for (int j = 0; j < 2; ++j) {
      float* cp = Cb + (long)(m0 + wm * 32 + i * 16 + 8 * half) * ldc
                     + n0 + wn * 32 + j * 16 + lm;
#pragma unroll
      for (int r = 0; r < 8; ++r) cp[(long)r * ldc] = acc[i][j][r];
    }
}

// ---------------------------------------------------------------------------
// Per-head-row (64 elems) RMSNorm in fp32, *weight, *scale, cast to bf16.
// One 32-lane wave per row, 8 rows per 256-thread block.
// Rows < s1 on the fused S axis use wC (c-stream weight), else wX.
// ---------------------------------------------------------------------------
__launch_bounds__(256)
__global__ void rmsnorm_to_bf16(const float* __restrict__ X, unsigned short* __restrict__ Y,
                                const float* __restrict__ wC, const float* __restrict__ wX,
                                int s1, int S, int H, float scale)
{
  const int t   = threadIdx.x;
  const int rid = blockIdx.x * 8 + (t >> 5);   // row = ((b*S + s)*H + h)
  const int idx = t & 31;
  const long base = (long)rid * 64;
  const int  s    = (rid / H) % S;

  float x0 = X[base + idx], x1 = X[base + idx + 32];
  float ss = x0 * x0 + x1 * x1;
#pragma unroll
  for (int off = 16; off; off >>= 1) ss += __shfl_xor(ss, off, 32);
  const float rr = rsqrtf(ss * (1.f / 64.f) + 1e-6f) * scale;
  const float* w = (s < s1) ? wC : wX;
  Y[base + idx]      = f2bf(x0 * rr * w[idx]);
  Y[base + idx + 32] = f2bf(x1 * rr * w[idx + 32]);
}

__launch_bounds__(256)
__global__ void cast_to_bf16(const float* __restrict__ X, unsigned short* __restrict__ Y)
{
  const long i = ((long)blockIdx.x * 256 + threadIdx.x) * 4;
  float4 v = *(const float4*)(X + i);
  uint2 p;
  p.x = pack2bf(v.x, v.y);
  p.y = pack2bf(v.z, v.w);
  *(uint2*)(Y + i) = p;
}

// ---------------------------------------------------------------------------
// Flash attention over the fused sequence with the block mask
//   allowed(q,k) = (k < s1) || (k <= q)
// 128 threads = 4 waves; 64 queries per block (16 per wave); key tiles of 32.
// Scores computed transposed (St[key][q]) so each lane owns one query column;
// softmax stats are per-lane + one shfl_xor(16) across the half split.
// Output accumulated transposed (Ot[d][q]) via Vt staged in LDS.
// Q/K/V are bf16 in [b, S, H, 64]; O is fp32 [b, S, H*64].
// ---------------------------------------------------------------------------
__launch_bounds__(128)
__global__ void flash_attn_wmma(const unsigned short* __restrict__ Q,
                                const unsigned short* __restrict__ Kmat,
                                const unsigned short* __restrict__ Vmat,
                                float* __restrict__ O,
                                int cLen, int S, int H)
{
  __shared__ unsigned short Ks[32 * 64];   // [key][d]
  __shared__ unsigned short Vs[64 * 32];   // [d][key]  (transposed V tile)

  const int qb = blockIdx.x * 64;
  const int h  = blockIdx.y;
  const int b  = blockIdx.z;
  const int t = threadIdx.x, wave = t >> 5, lane = t & 31, lm = lane & 15, half = lane >> 4;
  const int qg = qb + wave * 16 + lm;          // this lane's query row
  const int kl = t >> 2, c0 = (t & 3) * 16;    // staging: 32 rows x 64, 16 elems/thread

  // Q^T fragment (B operand): n = query = lane&15, k = head dim
  const unsigned short* qrow = Q + ((long)(b * S + qg) * H + h) * 64;
  ABFrag qf[2];
#pragma unroll
  for (int ks = 0; ks < 2; ++ks)
#pragma unroll
    for (int v = 0; v < 8; ++v)
      qf[ks].u[v] = *(const unsigned int*)&qrow[((v < 4) ? 2 * v : 16 + 2 * (v - 4)) + 8 * half + 32 * ks];

  f32x8 ot[4];
#pragma unroll
  for (int c = 0; c < 4; ++c) ot[c] = (f32x8){0.f,0.f,0.f,0.f,0.f,0.f,0.f,0.f};
  float m = -3.0e38f, l = 0.f;

  // s1 and 64-query tiles are 32-aligned -> every key tile is full.
  const int kmax = (qb + 64 <= cLen) ? cLen : (qb + 64);

  for (int kt = 0; kt < kmax; kt += 32) {
    __syncthreads();
    {
      const unsigned short* krow = Kmat + ((long)(b * S + kt + kl) * H + h) * 64 + c0;
      const unsigned short* vrow = Vmat + ((long)(b * S + kt + kl) * H + h) * 64 + c0;
      if (kt + 32 < kmax) {                  // warm L2 for the next key tile
        __builtin_prefetch(krow + (long)32 * H * 64, 0, 0);
        __builtin_prefetch(vrow + (long)32 * H * 64, 0, 0);
      }
      *(uint4*)&Ks[kl * 64 + c0]     = *(const uint4*)krow;
      *(uint4*)&Ks[kl * 64 + c0 + 8] = *(const uint4*)(krow + 8);

      unsigned short tmp[16] __attribute__((aligned(16)));
      *(uint4*)&tmp[0] = *(const uint4*)vrow;
      *(uint4*)&tmp[8] = *(const uint4*)(vrow + 8);
#pragma unroll
      for (int i = 0; i < 16; ++i) Vs[(c0 + i) * 32 + kl] = tmp[i];
    }
    __syncthreads();

    // St = K_tile (16x64) x Q^T (64x16), two key halves, two K-steps each
    f32x8 sA = {0.f,0.f,0.f,0.f,0.f,0.f,0.f,0.f};
    f32x8 sB = {0.f,0.f,0.f,0.f,0.f,0.f,0.f,0.f};
#pragma unroll
    for (int ks = 0; ks < 2; ++ks) {
      ABFrag k0, k1;
#pragma unroll
      for (int v = 0; v < 8; ++v) {
        const int kk = ((v < 4) ? 2 * v : 16 + 2 * (v - 4)) + 8 * half + 32 * ks;
        k0.u[v] = *(const unsigned int*)&Ks[lm * 64 + kk];
        k1.u[v] = *(const unsigned int*)&Ks[(16 + lm) * 64 + kk];
      }
      sA = __builtin_amdgcn_wmma_f32_16x16x32_bf16(false, k0.v, false, qf[ks].v,
                                                   (short)0, sA, false, false);
      sB = __builtin_amdgcn_wmma_f32_16x16x32_bf16(false, k1.v, false, qf[ks].v,
                                                   (short)0, sB, false, false);
    }

    // Mask + online softmax (scale 1/sqrt(64) was folded into Q at norm time)
    float p0[8], p1[8], lmax = -3.0e38f;
#pragma unroll
    for (int r = 0; r < 8; ++r) {
      const int kg0 = kt + r + 8 * half;
      const int kg1 = kg0 + 16;
      p0[r] = (kg0 < cLen || kg0 <= qg) ? sA[r] : -3.0e38f;
      p1[r] = (kg1 < cLen || kg1 <= qg) ? sB[r] : -3.0e38f;
      lmax = fmaxf(lmax, fmaxf(p0[r], p1[r]));
    }
    lmax = fmaxf(lmax, __shfl_xor(lmax, 16, 32));
    const float mnew  = fmaxf(m, lmax);
    const float alpha = __expf(m - mnew);
    float lsum = 0.f;
#pragma unroll
    for (int r = 0; r < 8; ++r) {
      p0[r] = __expf(p0[r] - mnew);
      p1[r] = __expf(p1[r] - mnew);
      lsum += p0[r] + p1[r];
    }
    l = l * alpha + lsum;
    m = mnew;
#pragma unroll
    for (int c = 0; c < 4; ++c) ot[c] *= alpha;

    // Repack P^T into the B-fragment layout (needs half-crossing exchange)
    float q0[8], q1[8];
#pragma unroll
    for (int r = 0; r < 8; ++r) {
      q0[r] = __shfl_xor(p0[r], 16, 32);
      q1[r] = __shfl_xor(p1[r], 16, 32);
    }
    ABFrag pb;
#pragma unroll
    for (int v = 0; v < 4; ++v) {
      const float x0 = half ? q1[2 * v]     : p0[2 * v];
      const float x1 = half ? q1[2 * v + 1] : p0[2 * v + 1];
      pb.u[v] = pack2bf(x0, x1);
      const float y0 = half ? p1[2 * v]     : q0[2 * v];
      const float y1 = half ? p1[2 * v + 1] : q0[2 * v + 1];
      pb.u[v + 4] = pack2bf(y0, y1);
    }

    // Ot[d][q] += V^T (16x32) x P^T (32x16) per 16-wide d chunk
#pragma unroll
    for (int c = 0; c < 4; ++c) {
      ABFrag va;
#pragma unroll
      for (int v = 0; v < 8; ++v) {
        const int kk = 2 * v + 16 * half;
        va.u[v] = *(const unsigned int*)&Vs[(c * 16 + lm) * 32 + kk];
      }
      ot[c] = __builtin_amdgcn_wmma_f32_16x16x32_bf16(false, va.v, false, pb.v,
                                                      (short)0, ot[c], false, false);
    }
  }

  const float inv = 1.f / (l + __shfl_xor(l, 16, 32));
  float* orow = O + (long)(b * S + qg) * (H * 64) + h * 64;
#pragma unroll
  for (int c = 0; c < 4; ++c)
#pragma unroll
    for (int r = 0; r < 8; ++r)
      orow[c * 16 + r + 8 * half] = ot[c][r] * inv;
}

// ---------------------------------------------------------------------------
extern "C" void kernel_launch(void* const* d_in, const int* in_sizes, int n_in,
                              void* d_out, int out_size, void* d_ws, size_t ws_size,
                              hipStream_t stream)
{
  (void)in_sizes; (void)n_in; (void)out_size; (void)ws_size;
  const int B = 2, S1 = 512, S2 = 2048, S = S1 + S2, DIM = 1024, H = 16;

  const float* c   = (const float*)d_in[0];
  const float* x   = (const float*)d_in[1];
  const float* w1q = (const float*)d_in[2];
  const float* w1k = (const float*)d_in[3];
  const float* w1v = (const float*)d_in[4];
  const float* w1o = (const float*)d_in[5];
  const float* w2q = (const float*)d_in[6];
  const float* w2k = (const float*)d_in[7];
  const float* w2v = (const float*)d_in[8];
  const float* w2o = (const float*)d_in[9];
  const float* qn1 = (const float*)d_in[10];
  const float* kn1 = (const float*)d_in[11];
  const float* qn2 = (const float*)d_in[12];
  const float* kn2 = (const float*)d_in[13];
  float* out = (float*)d_out;

  const long nTok = (long)B * S * DIM;                 // 5,242,880
  float* Qf = (float*)d_ws;                            // fp32 QKV intermediates
  float* Kf = Qf + nTok;
  float* Vf = Kf + nTok;
  unsigned short* Qb = (unsigned short*)(Vf + nTok);   // bf16 QKV for attention
  unsigned short* Kb = Qb + nTok;
  unsigned short* Vb = Kb + nTok;
  float* Of = Qf;                                      // reuse Qf after norm(Q)

  const long sAc = (long)S1 * DIM;
  const long sAx = (long)S2 * DIM;
  const long sQ  = (long)S  * DIM;
  dim3 blk(256);

  // QKV projections: c rows -> [0,512), x rows -> [512,2560) of the fused S axis
  gemm_bf16_wmma<<<dim3(16,  4, 2), blk, 0, stream>>>(c, sAc, DIM, w1q, DIM, Qf,                 sQ, DIM, DIM);
  gemm_bf16_wmma<<<dim3(16,  4, 2), blk, 0, stream>>>(c, sAc, DIM, w1k, DIM, Kf,                 sQ, DIM, DIM);
  gemm_bf16_wmma<<<dim3(16,  4, 2), blk, 0, stream>>>(c, sAc, DIM, w1v, DIM, Vf,                 sQ, DIM, DIM);
  gemm_bf16_wmma<<<dim3(16, 16, 2), blk, 0, stream>>>(x, sAx, DIM, w2q, DIM, Qf + (long)S1 * DIM, sQ, DIM, DIM);
  gemm_bf16_wmma<<<dim3(16, 16, 2), blk, 0, stream>>>(x, sAx, DIM, w2k, DIM, Kf + (long)S1 * DIM, sQ, DIM, DIM);
  gemm_bf16_wmma<<<dim3(16, 16, 2), blk, 0, stream>>>(x, sAx, DIM, w2v, DIM, Vf + (long)S1 * DIM, sQ, DIM, DIM);

  // RMSNorm Q (fold 1/sqrt(hd)=0.125 into Q), RMSNorm K, plain cast V
  const int rowsBlocks = (B * S * H) / 8;
  rmsnorm_to_bf16<<<dim3(rowsBlocks), blk, 0, stream>>>(Qf, Qb, qn1, qn2, S1, S, H, 0.125f);
  rmsnorm_to_bf16<<<dim3(rowsBlocks), blk, 0, stream>>>(Kf, Kb, kn1, kn2, S1, S, H, 1.0f);
  cast_to_bf16<<<dim3((unsigned)(nTok / 1024)), blk, 0, stream>>>(Vf, Vb);

  // Attention over the fused sequence
  flash_attn_wmma<<<dim3(S / 64, H, B), dim3(128), 0, stream>>>(Qb, Kb, Vb, Of, S1, S, H);

  // Output projections: c part through w1o, x part through w2o
  gemm_bf16_wmma<<<dim3(16,  4, 2), blk, 0, stream>>>(Of,                 sQ, DIM, w1o, DIM, out,                     (long)S1 * DIM, DIM, DIM);
  gemm_bf16_wmma<<<dim3(16, 16, 2), blk, 0, stream>>>(Of + (long)S1 * DIM, sQ, DIM, w2o, DIM, out + (long)B * S1 * DIM, (long)S2 * DIM, DIM, DIM);
}
